// MultiHeadSelfAttention_70875550319030
// MI455X (gfx1250) — matrix operand order repaired and checked by reading, not compiled, not run on hardware
//
#include <hip/hip_runtime.h>
#include <hip/hip_bf16.h>

#define HIDDEN 1024
#define HEADS 16
#define HEAD_DIM 64
#define PREFIX 16
#define BATCH 2
#define SEQ 2048
#define S_TOT 2064
#define S_PAD 2080   // 65 * 32

typedef __attribute__((ext_vector_type(16))) __bf16 v16bf;
typedef __attribute__((ext_vector_type(8)))  float  v8f;
typedef __attribute__((ext_vector_type(8)))  unsigned short u16x8;
typedef __attribute__((ext_vector_type(16))) unsigned short u16x16;
typedef __attribute__((ext_vector_type(4)))  unsigned int   u32x4;
typedef __attribute__((ext_vector_type(8)))  int            i32x8;
typedef __attribute__((ext_vector_type(4)))  int            i32x4;

#define TDM_OK __has_builtin(__builtin_amdgcn_tensor_load_to_lds)

__device__ __forceinline__ unsigned short f2bf(float f) {
  unsigned int x = __builtin_bit_cast(unsigned int, f);
  x += 0x7FFFu + ((x >> 16) & 1u);
  return (unsigned short)(x >> 16);
}

__device__ __forceinline__ u16x16 cat8(u16x8 a, u16x8 b) {
  return __builtin_shufflevector(a, b, 0,1,2,3,4,5,6,7,8,9,10,11,12,13,14,15);
}

__device__ __forceinline__ v8f wmma_bf16(u16x16 a, u16x16 b, v8f c) {
  return __builtin_amdgcn_wmma_f32_16x16x32_bf16(
      false, __builtin_bit_cast(v16bf, a),
      false, __builtin_bit_cast(v16bf, b),
      (short)0, c, false, false);
}

__device__ __forceinline__ v8f v8f_zero() {
  v8f z;
#pragma unroll
  for (int i = 0; i < 8; ++i) z[i] = 0.0f;
  return z;
}

#if TDM_OK
// 2D TDM: load 128 rows x 32 cols of bf16 (row stride = stride_elems) into LDS.
// D# per CDNA5 ISA (group0: count/lds/global/type,
// group1: data_size=2B, big tensor dims (no OOB), tile 32x128, dim0 stride).
// Toolchain here is the 6-arg builtin (clang-23/therock form).
__device__ __forceinline__ void tdm_load_tile(const unsigned short* gptr,
                                              unsigned lds_off, int stride_elems) {
  unsigned long long ga = (unsigned long long)(const void*)gptr;
  u32x4 g0;
  g0[0] = 1u;                                                  // count=1
  g0[1] = lds_off;                                             // LDS byte addr
  g0[2] = (unsigned)ga;                                        // global lo
  g0[3] = ((unsigned)(ga >> 32) & 0x01FFFFFFu) | 0x80000000u;  // global hi | type=2
  i32x8 g1;
  g1[0] = 0x00010000;            // data_size = 1 (2 bytes)
  g1[1] = 0;                     // barrier addr 0 | tensor_dim0[15:0]=0
  g1[2] = 0x00000010;            // tensor_dim0 = 0x100000 ; dim1[15:0]=0
  g1[3] = (32 << 16) | 0x10;     // tensor_dim1 = 0x100000 ; tile_dim0 = 32
  g1[4] = 128;                   // tile_dim1 = 128 (tile_dim2 = 0 -> 2D)
  g1[5] = stride_elems;          // tensor_dim0_stride[31:0]
  g1[6] = 0;                     // stride hi | dim1_stride lo
  g1[7] = 0;
  i32x4 z4 = {0, 0, 0, 0};
  i32x8 z8 = {0, 0, 0, 0, 0, 0, 0, 0};
  __builtin_amdgcn_tensor_load_to_lds(g0, g1, z4, z4, z8, 0);
}
#endif

// ---------------------------------------------------------------------------
// fp32 -> bf16 convert (plain)
// ---------------------------------------------------------------------------
__global__ void cvt_f32_bf16_kernel(const float* __restrict__ src,
                                    unsigned short* __restrict__ dst, int n) {
  int i = blockIdx.x * blockDim.x + threadIdx.x;
  if (i < n) dst[i] = f2bf(src[i]);
}

// ---------------------------------------------------------------------------
// fp32 [K][N] -> bf16 transposed [N][K], 32x32 LDS tiles (coalesced both ways)
// ---------------------------------------------------------------------------
__global__ __launch_bounds__(256) void cvt_transpose_kernel(
    const float* __restrict__ src, unsigned short* __restrict__ dst, int K, int N) {
  __shared__ float tile[32][33];
  const int kb = blockIdx.y * 32, nb = blockIdx.x * 32;
  const int tx = threadIdx.x & 31, ty = threadIdx.x >> 5;  // 32 x 8
#pragma unroll
  for (int r = ty; r < 32; r += 8)
    tile[r][tx] = src[(size_t)(kb + r) * N + nb + tx];
  __syncthreads();
#pragma unroll
  for (int r = ty; r < 32; r += 8)
    dst[(size_t)(nb + r) * K + kb + tx] = f2bf(tile[tx][r]);
}

// ---------------------------------------------------------------------------
// Prefix K/V copy + zero pad rows 2064..2079.
// K: [B,H,S_PAD,D]  (s-major)   V: [B,H,D,S_PAD]  (d-major, transposed)
// idx bits: d:6 rr:5 h:4 b:1 -> 65536 threads
// ---------------------------------------------------------------------------
__global__ void prefix_pad_kernel(const float* __restrict__ pk,
                                  const float* __restrict__ pv,
                                  unsigned short* __restrict__ Kb,
                                  unsigned short* __restrict__ Vt) {
  int idx = blockIdx.x * blockDim.x + threadIdx.x;
  if (idx >= BATCH * HEADS * 32 * HEAD_DIM) return;
  int d  = idx & 63;
  int rr = (idx >> 6) & 31;
  int h  = (idx >> 11) & 15;
  int b  = idx >> 15;
  size_t bh = (size_t)(b * HEADS + h);
  int s = (rr < PREFIX) ? rr : (S_TOT + (rr - 16));
  unsigned short kv = 0, vv = 0;
  if (rr < PREFIX) {
    size_t src = ((size_t)h * PREFIX + rr) * HEAD_DIM + d;  // broadcast over b
    kv = f2bf(pk[src]);
    vv = f2bf(pv[src]);
  }
  Kb[(bh * S_PAD + s) * HEAD_DIM + d]  = kv;
  Vt[(bh * HEAD_DIM + d) * S_PAD + s]  = vv;
}

// ---------------------------------------------------------------------------
// GEMM tile engine: 256 threads (8 waves), 128x128 tile, K step 32.
// A row-major [M][K]; Bt row-major [N][K] (weights pre-transposed).
// TDM double-buffered staging (wave 0 issues, s_wait_tensorcnt pipelines),
// fallback: manual b128 staging. Fragments are contiguous ds_load_b128.
// ---------------------------------------------------------------------------
__device__ __forceinline__ void gemm_tiles(const unsigned short* __restrict__ A,
                                           const unsigned short* __restrict__ Bt,
                                           int lda, int ldbt, int K,
                                           int mbase, int nbase,
                                           unsigned short (*sA)[128 * 32],
                                           unsigned short (*sB)[128 * 32],
                                           v8f acc[2][4]) {
  const int tid  = threadIdx.x;
  const int lane = tid & 31;
  const int w    = tid >> 5;
  const int wm   = w & 3, wn = w >> 2;
  const int hf   = lane >> 4, ln = lane & 15;
  const int base0 = hf * 8;      // A-layout K sub-base
  const int kbase = hf * 16;     // unused (B fragments use base0 as well)
  const int nIter = K / 32;

#if TDM_OK
  const bool issuer = (__builtin_amdgcn_readfirstlane(tid) >> 5) == 0;  // scalar branch
  const unsigned ldsA = (unsigned)(unsigned long long)(void*)&sA[0][0];
  const unsigned ldsB = (unsigned)(unsigned long long)(void*)&sB[0][0];
  const unsigned BUFB = 128 * 32 * 2;   // bytes per buffer
  if (issuer) {
    tdm_load_tile(A  + (size_t)mbase * lda,  ldsA, lda);
    tdm_load_tile(Bt + (size_t)nbase * ldbt, ldsB, ldbt);
  }
#endif

  for (int it = 0; it < nIter; ++it) {
    const int kk  = it * 32;
    const int buf =
#if TDM_OK
        it & 1;
    if (issuer && (it + 1 < nIter)) {   // prefetch next tile into other buffer
      tdm_load_tile(A  + (size_t)mbase * lda  + kk + 32, ldsA + (buf ^ 1) * BUFB, lda);
      tdm_load_tile(Bt + (size_t)nbase * ldbt + kk + 32, ldsB + (buf ^ 1) * BUFB, ldbt);
    }
    if (it + 1 < nIter) __builtin_amdgcn_s_wait_tensorcnt(2);
    else                __builtin_amdgcn_s_wait_tensorcnt(0);
    __syncthreads();
#else
        0;
    {  // manual staging: both tiles are contiguous 16B load/store pairs
      int r  = tid >> 1;
      int cg = (tid & 1) * 16;
      const u16x8* ga = (const u16x8*)(A + (size_t)(mbase + r) * lda + kk + cg);
      u16x8 a0 = ga[0], a1 = ga[1];
      *(u16x8*)(&sA[0][0] + r * 32 + cg)     = a0;
      *(u16x8*)(&sA[0][0] + r * 32 + cg + 8) = a1;
      const u16x8* gb = (const u16x8*)(Bt + (size_t)(nbase + r) * ldbt + kk + cg);
      u16x8 b0 = gb[0], b1 = gb[1];
      *(u16x8*)(&sB[0][0] + r * 32 + cg)     = b0;
      *(u16x8*)(&sB[0][0] + r * 32 + cg + 8) = b1;
    }
    __syncthreads();
#endif

    const unsigned short* cA = &sA[buf][0];
    const unsigned short* cB = &sB[buf][0];
    u16x16 af[2];
#pragma unroll
    for (int mf = 0; mf < 2; ++mf) {
      int m = wm * 32 + mf * 16 + ln;   // lane row = L%16 (A layout)
      af[mf] = cat8(*(const u16x8*)(cA + m * 32 + base0),
                    *(const u16x8*)(cA + m * 32 + 16 + base0));
    }
#pragma unroll
    for (int nf = 0; nf < 4; ++nf) {
      int n = wn * 64 + nf * 16 + ln;   // lane col = L%16 (B layout)
      u16x16 bf = cat8(*(const u16x8*)(cB + n * 32 + base0),
                       *(const u16x8*)(cB + n * 32 + 16 + base0));
#pragma unroll
      for (int mf = 0; mf < 2; ++mf)
        acc[mf][nf] = wmma_bf16(af[mf], bf, acc[mf][nf]);
    }
    __syncthreads();   // protect buffer reuse before next DMA/stage overwrites
  }
}

// ---------------------------------------------------------------------------
// GEMM 1: qkv = Xb[4096x1024] @ Wqkv_t[3072x1024]^T + b ; scatter bf16 Q/K/V
// ---------------------------------------------------------------------------
__global__ __launch_bounds__(256) void qkv_gemm_kernel(
    const unsigned short* __restrict__ Xb, const unsigned short* __restrict__ Wbt,
    const float* __restrict__ bias,
    unsigned short* __restrict__ Qb, unsigned short* __restrict__ Kb,
    unsigned short* __restrict__ Vt) {
  __shared__ unsigned short sA[2][128 * 32];
  __shared__ unsigned short sB[2][128 * 32];
  const int mbase = blockIdx.y * 128;
  const int nbase = blockIdx.x * 128;
  v8f acc[2][4];
#pragma unroll
  for (int a = 0; a < 2; ++a)
#pragma unroll
    for (int b = 0; b < 4; ++b) acc[a][b] = v8f_zero();

  gemm_tiles(Xb, Wbt, HIDDEN, HIDDEN, HIDDEN, mbase, nbase, sA, sB, acc);

  const int lane = threadIdx.x & 31;
  const int w    = threadIdx.x >> 5;
  const int wm = w & 3, wn = w >> 2, hf = lane >> 4, ln = lane & 15;
#pragma unroll
  for (int nf = 0; nf < 4; ++nf) {
    int n = nbase + wn * 64 + nf * 16 + ln;
    float bv = bias[n];
    int qid = n >> 10, rem = n & 1023;
    int h = rem >> 6, d = rem & 63;
#pragma unroll
    for (int mf = 0; mf < 2; ++mf) {
#pragma unroll
      for (int i = 0; i < 8; ++i) {
        int m = mbase + wm * 32 + mf * 16 + i + 8 * hf;   // C layout row
        int b = m >> 11, t = m & 2047;
        unsigned short bf = f2bf(acc[mf][nf][i] + bv);
        size_t bh = (size_t)(b * HEADS + h);
        if (qid == 0)      Qb[(bh * SEQ + t) * HEAD_DIM + d] = bf;
        else if (qid == 1) Kb[(bh * S_PAD + PREFIX + t) * HEAD_DIM + d] = bf;
        else               Vt[(bh * HEAD_DIM + d) * S_PAD + PREFIX + t] = bf;
      }
    }
  }
}

// ---------------------------------------------------------------------------
// Flash attention. 1 block = (b,h, 64 query rows); 4 waves x 16 rows.
// K s-major (B-frag contiguous in d), V d-major (B-frag contiguous in s):
// both operands load straight from global as b128 pairs; LDS only for the
// P C-layout -> A-layout round trip.
// ---------------------------------------------------------------------------
__global__ __launch_bounds__(128) void attn_kernel(
    const unsigned short* __restrict__ Qb, const unsigned short* __restrict__ Kb,
    const unsigned short* __restrict__ Vt, unsigned short* __restrict__ Ctx) {
  __shared__ unsigned short sP[4 * 16 * 32];   // per-wave P tile [16][32]

  const int tid  = threadIdx.x;
  const int lane = tid & 31;
  const int w    = tid >> 5;
  const int hf   = lane >> 4, ln = lane & 15;
  const int base0 = hf * 8, kbase = hf * 16;
  const int bh   = blockIdx.x >> 5;        // b*H + h
  const int tblk = blockIdx.x & 31;
  const int t0   = tblk * 64 + w * 16;
  const size_t koff = (size_t)bh * S_PAD * HEAD_DIM;
  const size_t voff = (size_t)bh * HEAD_DIM * S_PAD;
  unsigned short* myP = sP + w * (16 * 32);

  // Q fragments (A layout): lane row = t0 + L%16, two contiguous 8-half runs
  u16x16 aq[2];
  const size_t qoff = ((size_t)bh * SEQ + t0 + ln) * HEAD_DIM;
#pragma unroll
  for (int df = 0; df < 2; ++df) {
    const unsigned short* p = Qb + qoff + df * 32 + base0;
    aq[df] = cat8(*(const u16x8*)p, *(const u16x8*)(p + 16));
  }

  float mrow[8], lrow[8];
  v8f oacc[4];
#pragma unroll
  for (int i = 0; i < 8; ++i) { mrow[i] = -3.0e38f; lrow[i] = 0.0f; }
#pragma unroll
  for (int dt = 0; dt < 4; ++dt) oacc[dt] = v8f_zero();

  for (int s0 = 0; s0 < S_PAD; s0 += 32) {
    if (s0 + 32 < S_PAD)   // hint next K chunk (global_prefetch_b8)
      __builtin_prefetch(Kb + koff + (size_t)(s0 + 32 + ln) * HEAD_DIM, 0, 1);

    // scores: two 16-col tiles, each accumulated over D=64 (2 x K32)
    v8f sc[2];
#pragma unroll
    for (int st = 0; st < 2; ++st) {
      sc[st] = v8f_zero();
      int s = s0 + st * 16 + ln;       // B-layout lane col
      const unsigned short* kp = Kb + koff + (size_t)s * HEAD_DIM + base0;
#pragma unroll
      for (int df = 0; df < 2; ++df) {
        u16x16 bk = cat8(*(const u16x8*)(kp + df * 32),
                         *(const u16x8*)(kp + df * 32 + 16));
        sc[st] = wmma_bf16(aq[df], bk, sc[st]);
      }
    }

    // scale + mask padded columns
#pragma unroll
    for (int st = 0; st < 2; ++st) {
      bool oob = (s0 + st * 16 + ln) >= S_TOT;
#pragma unroll
      for (int i = 0; i < 8; ++i) {
        float v = sc[st][i] * 0.125f;   // 1/sqrt(64)
        sc[st][i] = oob ? -3.0e38f : v;
      }
    }

    // online softmax (row stats replicated across each 16-lane half)
    float corr[8];
#pragma unroll
    for (int i = 0; i < 8; ++i) {
      float cm = fmaxf(sc[0][i], sc[1][i]);
      cm = fmaxf(cm, __shfl_xor(cm, 1, 16));
      cm = fmaxf(cm, __shfl_xor(cm, 2, 16));
      cm = fmaxf(cm, __shfl_xor(cm, 4, 16));
      cm = fmaxf(cm, __shfl_xor(cm, 8, 16));
      float mn = fmaxf(mrow[i], cm);
      float p0 = __expf(sc[0][i] - mn);
      float p1 = __expf(sc[1][i] - mn);
      sc[0][i] = p0; sc[1][i] = p1;
      float rs = p0 + p1;
      rs += __shfl_xor(rs, 1, 16);
      rs += __shfl_xor(rs, 2, 16);
      rs += __shfl_xor(rs, 4, 16);
      rs += __shfl_xor(rs, 8, 16);
      float c = __expf(mrow[i] - mn);
      lrow[i] = lrow[i] * c + rs;
      mrow[i] = mn;
      corr[i] = c;
    }
#pragma unroll
    for (int dt = 0; dt < 4; ++dt)
#pragma unroll
      for (int i = 0; i < 8; ++i) oacc[dt][i] *= corr[i];

    // P (C layout) -> LDS -> reload as A-layout fragment
    __syncthreads();
#pragma unroll
    for (int i = 0; i < 8; ++i) {
      int row = i + 8 * hf;
      myP[row * 32 + ln]      = f2bf(sc[0][i]);
      myP[row * 32 + 16 + ln] = f2bf(sc[1][i]);
    }
    __syncthreads();
    u16x16 ap = cat8(*(const u16x8*)(myP + ln * 32 + base0),
                     *(const u16x8*)(myP + ln * 32 + 16 + base0));

    // PV: 4 d-tiles; V fragment (B layout, contiguous in s) from global
#pragma unroll
    for (int dt = 0; dt < 4; ++dt) {
      const unsigned short* vp = Vt + voff + (size_t)(dt * 16 + ln) * S_PAD + s0 + base0;
      u16x16 bv = cat8(*(const u16x8*)vp, *(const u16x8*)(vp + 16));
      oacc[dt] = wmma_bf16(ap, bv, oacc[dt]);
    }
  }

  // normalize and write ctx [B*T, H*D] bf16
  const int bb = bh >> 4, h = bh & 15;
#pragma unroll
  for (int i = 0; i < 8; ++i) {
    float inv = 1.0f / lrow[i];
    int tq = t0 + i + 8 * hf;
    size_t rowoff = ((size_t)bb * SEQ + tq) * HIDDEN + h * HEAD_DIM;
#pragma unroll
    for (int dt = 0; dt < 4; ++dt)
      Ctx[rowoff + dt * 16 + ln] = f2bf(oacc[dt][i] * inv);
  }
}

// ---------------------------------------------------------------------------
// GEMM 2: out = Ctx[4096x1024] @ Wout_t[1024x1024]^T + b_out  (fp32 out)
// ---------------------------------------------------------------------------
__global__ __launch_bounds__(256) void out_gemm_kernel(
    const unsigned short* __restrict__ Ctx, const unsigned short* __restrict__ Wbt,
    const float* __restrict__ bias, float* __restrict__ out) {
  __shared__ unsigned short sA[2][128 * 32];
  __shared__ unsigned short sB[2][128 * 32];
  const int mbase = blockIdx.y * 128;
  const int nbase = blockIdx.x * 128;
  v8f acc[2][4];
#pragma unroll
  for (int a = 0; a < 2; ++a)
#pragma unroll
    for (int b = 0; b < 4; ++b) acc[a][b] = v8f_zero();

  gemm_tiles(Ctx, Wbt, HIDDEN, HIDDEN, HIDDEN, mbase, nbase, sA, sB, acc);

  const int lane = threadIdx.x & 31;
  const int w    = threadIdx.x >> 5;
  const int wm = w & 3, wn = w >> 2, hf = lane >> 4, ln = lane & 15;
#pragma unroll
  for (int nf = 0; nf < 4; ++nf) {
    int n = nbase + wn * 64 + nf * 16 + ln;
    float bv = bias[n];
#pragma unroll
    for (int mf = 0; mf < 2; ++mf) {
#pragma unroll
      for (int i = 0; i < 8; ++i) {
        int m = mbase + wm * 32 + mf * 16 + i + 8 * hf;
        out[(size_t)m * HIDDEN + n] = acc[mf][nf][i] + bv;
      }
    }
  }
}

// ---------------------------------------------------------------------------
extern "C" void kernel_launch(void* const* d_in, const int* in_sizes, int n_in,
                              void* d_out, int out_size, void* d_ws, size_t ws_size,
                              hipStream_t stream) {
  (void)in_sizes; (void)n_in; (void)out_size; (void)ws_size;
  const float* x    = (const float*)d_in[0];
  const float* pk   = (const float*)d_in[1];
  const float* pv   = (const float*)d_in[2];
  const float* Wqkv = (const float*)d_in[3];
  const float* bqkv = (const float*)d_in[4];
  const float* Wout = (const float*)d_in[5];
  const float* bout = (const float*)d_in[6];
  float* out = (float*)d_out;

  char* ws = (char*)d_ws;
  size_t off = 0;
  auto carve = [&](size_t bytes) -> unsigned short* {
    unsigned short* p = (unsigned short*)(ws + off);
    off = (off + bytes + 255) & ~(size_t)255;
    return p;
  };
  const size_t M = (size_t)BATCH * SEQ;                          // 4096
  unsigned short* Xb   = carve(M * HIDDEN * 2);                  // x bf16
  unsigned short* Wqbt = carve((size_t)HIDDEN * 3 * HIDDEN * 2); // Wqkv^T bf16
  unsigned short* Wobt = carve((size_t)HIDDEN * HIDDEN * 2);     // Wout^T bf16
  unsigned short* Qb   = carve((size_t)BATCH * HEADS * SEQ * HEAD_DIM * 2);
  unsigned short* Kb   = carve((size_t)BATCH * HEADS * S_PAD * HEAD_DIM * 2);
  unsigned short* Vt   = carve((size_t)BATCH * HEADS * HEAD_DIM * S_PAD * 2);
  unsigned short* Ctx  = carve(M * HIDDEN * 2);

  // 1) converts (x plain; weights transposed so GEMM B-tiles are contiguous)
  {
    int n = (int)(M * HIDDEN);
    cvt_f32_bf16_kernel<<<(n + 255) / 256, 256, 0, stream>>>(x, Xb, n);
  }
  cvt_transpose_kernel<<<dim3(3 * HIDDEN / 32, HIDDEN / 32), 256, 0, stream>>>(
      Wqkv, Wqbt, HIDDEN, 3 * HIDDEN);
  cvt_transpose_kernel<<<dim3(HIDDEN / 32, HIDDEN / 32), 256, 0, stream>>>(
      Wout, Wobt, HIDDEN, HIDDEN);
  // 2) prefix + pad
  prefix_pad_kernel<<<(BATCH * HEADS * 32 * HEAD_DIM) / 256, 256, 0, stream>>>(
      pk, pv, Kb, Vt);
  // 3) QKV GEMM (N=3072 -> 24 tiles, M=4096 -> 32 tiles)
  qkv_gemm_kernel<<<dim3(3 * HIDDEN / 128, M / 128), 256, 0, stream>>>(
      Xb, Wqbt, bqkv, Qb, Kb, Vt);
  // 4) attention: B*H*(T/64) = 1024 blocks of 128 threads
  attn_kernel<<<BATCH * HEADS * (SEQ / 64), 128, 0, stream>>>(Qb, Kb, Vt, Ctx);
  // 5) out GEMM
  out_gemm_kernel<<<dim3(HIDDEN / 128, M / 128), 256, 0, stream>>>(
      Ctx, Wobt, bout, out);
}
